// MemNet_46411416600664
// MI455X (gfx1250) — matrix operand order repaired and checked by reading, not compiled, not run on hardware
//
#include <hip/hip_runtime.h>
#include <hip/hip_bf16.h>
#include <math.h>
#include <stdint.h>

typedef __attribute__((ext_vector_type(16))) __bf16 v16bf;
typedef __attribute__((ext_vector_type(8)))  float  v8f;

#define NB    128        // batch
#define OBSD  64
#define ACTD  16
#define RETD  8
#define MEMD  88
#define NMEM  1000000
#define TOPK  16
#define NCH   128        // memory chunks (== grid of phase-1)
#define EMB   64
#define OUTD  64
#define NCAND (NCH * TOPK)   // 2048 candidates per query

#define NEG_INF (-__builtin_inff())

// LDS byte offset of a generic pointer to __shared__ (low 32 bits of the address)
#define LDS_OFF(p) ((unsigned)(uintptr_t)(p))

// ---------------------------------------------------------------------------
// Phase 1: streaming similarity scan (bf16 WMMA) + per-chunk top-16 per query.
// Grid: NCH blocks x 256 threads (8 waves). Each block scans one memory chunk
// (32 rows per iteration) for ALL 128 queries; wave w owns queries [16w,16w+16).
// The loader pre-scales each memory row by rsqrt(|row|^2) before the bf16
// conversion, so the WMMA accumulator directly yields the cosine score.
// ---------------------------------------------------------------------------
__global__ __launch_bounds__(256)
void p1_scores_topk(const float* __restrict__ obs,
                    const float* __restrict__ memories,
                    float* __restrict__ part_score,
                    int*   __restrict__ part_idx)
{
    __shared__ __align__(32) __bf16 tileB[32][64];   // 32 normalized rows (bf16)
    __shared__ __align__(16) float stag[8][16][32];  // per-wave score tile [q][col]
    __shared__ float lsc[8][32][16];                 // per-lane top-16 scores
    __shared__ int   lix[8][32][16];                 // per-lane top-16 indices

    const int tid  = threadIdx.x;
    const int wave = tid >> 5;
    const int lane = tid & 31;
    const int lo   = lane & 15;
    const bool hiH = (lane >= 16);

    // init candidate lists to -inf
    {
        float* ls = &lsc[0][0][0];
        int*   li = &lix[0][0][0];
        for (int i = tid; i < 8 * 32 * 16; i += 256) { ls[i] = NEG_INF; li[i] = 0; }
    }

    // A fragments: this wave's 16 queries, K = 64 (two 16x32 bf16 fragments)
    const int q0 = wave * 16;
    v16bf a0, a1;
    {
        const float* orow = obs + (size_t)(q0 + lo) * OBSD;
        const int kb = hiH ? 16 : 0;
        #pragma unroll
        for (int e = 0; e < 8; ++e) {
            a0[e]     = (__bf16)orow[kb + e];
            a0[e + 8] = (__bf16)orow[8 + kb + e];
            a1[e]     = (__bf16)orow[32 + kb + e];
            a1[e + 8] = (__bf16)orow[40 + kb + e];
        }
    }

    const int chunk = (NMEM + NCH - 1) / NCH;        // 7813
    const int c0 = blockIdx.x * chunk;
    int c1 = c0 + chunk; if (c1 > NMEM) c1 = NMEM;

    float thresh = NEG_INF;                          // min of my 16-entry list
    int   minpos = 0;

    const int lrow = tid >> 3;                       // loader: row 0..31
    const int lseg = tid & 7;                        // loader: 8-float segment 0..7

    for (int t0 = c0; t0 < c1; t0 += 32) {
        const bool tail = (t0 + 32 > c1);            // uniform scalar
        __syncthreads();                             // tile reuse fence
        // cooperative load: 32 rows -> normalized bf16 tile
        {
            float4 va, vb;
            if (!tail) {                             // steady state: unguarded
                const float4* p = reinterpret_cast<const float4*>(
                    memories + (size_t)(t0 + lrow) * MEMD);
                va = p[lseg * 2];
                vb = p[lseg * 2 + 1];
                if (t0 + 160 <= c1)                  // uniform prefetch guard
                    __builtin_prefetch(memories + (size_t)(t0 + 128 + lrow) * MEMD + lseg * 8, 0, 0);
            } else {
                va = make_float4(0.f, 0.f, 0.f, 0.f);
                vb = va;
                if (t0 + lrow < c1) {
                    const float4* p = reinterpret_cast<const float4*>(
                        memories + (size_t)(t0 + lrow) * MEMD);
                    va = p[lseg * 2];
                    vb = p[lseg * 2 + 1];
                }
            }
            float part = va.x * va.x + va.y * va.y + va.z * va.z + va.w * va.w
                       + vb.x * vb.x + vb.y * vb.y + vb.z * vb.z + vb.w * vb.w;
            // deterministic xor-tree reduce across the 8 loader lanes of this row
            part += __shfl_xor(part, 1);
            part += __shfl_xor(part, 2);
            part += __shfl_xor(part, 4);
            // 1/max(sqrt(x),1e-12) == rsqrt(max(x,1e-24)) -> single v_rsq_f32
            const float invn = __builtin_amdgcn_rsqf(fmaxf(part, 1e-24f));
            __bf16* dst = &tileB[lrow][lseg * 8];
            dst[0] = (__bf16)(va.x * invn); dst[1] = (__bf16)(va.y * invn);
            dst[2] = (__bf16)(va.z * invn); dst[3] = (__bf16)(va.w * invn);
            dst[4] = (__bf16)(vb.x * invn); dst[5] = (__bf16)(vb.y * invn);
            dst[6] = (__bf16)(vb.z * invn); dst[7] = (__bf16)(vb.w * invn);
        }
        __syncthreads();                             // tile ready

        // per-wave WMMA: 16 queries x 32 memory rows, K=64 -> final scores
        const int koff = hiH ? 16 : 0;
        v16bf b00 = *reinterpret_cast<const v16bf*>(&tileB[lo][koff]);
        v16bf b01 = *reinterpret_cast<const v16bf*>(&tileB[lo][32 + koff]);
        v16bf b10 = *reinterpret_cast<const v16bf*>(&tileB[16 + lo][koff]);
        v16bf b11 = *reinterpret_cast<const v16bf*>(&tileB[16 + lo][32 + koff]);
        v8f ca = {}, cb = {};
        ca = __builtin_amdgcn_wmma_f32_16x16x32_bf16(false, a0, false, b00, (short)0, ca, false, false);
        ca = __builtin_amdgcn_wmma_f32_16x16x32_bf16(false, a1, false, b01, (short)0, ca, false, false);
        cb = __builtin_amdgcn_wmma_f32_16x16x32_bf16(false, a0, false, b10, (short)0, cb, false, false);
        cb = __builtin_amdgcn_wmma_f32_16x16x32_bf16(false, a1, false, b11, (short)0, cb, false, false);

        const int mrow = hiH ? 8 : 0;
        if (!tail) {                                 // steady state: no masking
            #pragma unroll
            for (int r = 0; r < 8; ++r) {
                stag[wave][mrow + r][lo]      = ca[r];
                stag[wave][mrow + r][16 + lo] = cb[r];
            }
        } else {
            const bool val0 = (t0 + lo) < c1;
            const bool val1 = (t0 + 16 + lo) < c1;
            #pragma unroll
            for (int r = 0; r < 8; ++r) {
                stag[wave][mrow + r][lo]      = val0 ? ca[r] : NEG_INF;
                stag[wave][mrow + r][16 + lo] = val1 ? cb[r] : NEG_INF;
            }
        }

        // staging is per-wave; same-wave DS ops are in-order -> wave fence only
        __builtin_amdgcn_wave_barrier();
        asm volatile("" ::: "memory");

        // top-16 insert: lane handles (query = lo, 16 cols starting at cbase)
        const int cbase = hiH ? 16 : 0;
        float ss[16];
        {
            const float4* sp = reinterpret_cast<const float4*>(&stag[wave][lo][cbase]);
            #pragma unroll
            for (int g = 0; g < 4; ++g) {
                const float4 x = sp[g];
                ss[g * 4 + 0] = x.x; ss[g * 4 + 1] = x.y;
                ss[g * 4 + 2] = x.z; ss[g * 4 + 3] = x.w;
            }
        }
        float mx = ss[0];
        #pragma unroll
        for (int j = 1; j < 16; ++j) mx = fmaxf(mx, ss[j]);
        if (mx > thresh) {                           // rare after warmup
            float* myS = &lsc[wave][lane][0];
            int*   myI = &lix[wave][lane][0];
            #pragma unroll
            for (int j = 0; j < 16; ++j) {
                if (ss[j] > thresh) {
                    myS[minpos] = ss[j];
                    myI[minpos] = t0 + cbase + j;
                    float mn = __builtin_inff(); int mp = 0;
                    #pragma unroll
                    for (int u = 0; u < 16; ++u) {
                        const float vv = myS[u];
                        if (vv < mn) { mn = vv; mp = u; }
                    }
                    thresh = mn; minpos = mp;
                }
            }
        }
    }

    __syncthreads();
    // merge the two 16-entry half-lists -> top-16, write chunk partials
    if (!hiH) {
        float* sA = &lsc[wave][lo][0];
        float* sB = &lsc[wave][lo + 16][0];
        int*   iA = &lix[wave][lo][0];
        int*   iB = &lix[wave][lo + 16][0];
        const int q = q0 + lo;
        float* outS = part_score + ((size_t)q * NCH + blockIdx.x) * TOPK;
        int*   outI = part_idx   + ((size_t)q * NCH + blockIdx.x) * TOPK;
        for (int k = 0; k < TOPK; ++k) {
            float best = NEG_INF; int bj = 0;
            for (int j = 0; j < 32; ++j) {
                const float v = (j < 16) ? sA[j] : sB[j - 16];
                if (v > best) { best = v; bj = j; }
            }
            outS[k] = best;
            outI[k] = (bj < 16) ? iA[bj] : iB[bj - 16];
            if (bj < 16) sA[bj] = NEG_INF; else sB[bj - 16] = NEG_INF;
        }
    }
}

// ---------------------------------------------------------------------------
// Phase 2: per query, merge 2048 chunk candidates -> global top-16, pick the
// one with the largest return-sum, emit its 16-dim action.
// Candidates are pulled global->LDS with GLOBAL_LOAD_ASYNC_TO_LDS_B128.
// Grid: NB blocks x 256 threads.
// ---------------------------------------------------------------------------
__global__ __launch_bounds__(256)
void p2_select(const float* __restrict__ memories,
               const float* __restrict__ part_score,
               const int*   __restrict__ part_idx,
               float* __restrict__ best_act)
{
    __shared__ __align__(16) float cs[NCAND];
    __shared__ __align__(16) int   ci[NCAND];
    __shared__ float rmax[256];
    __shared__ int   rpos[256];
    __shared__ int   win[TOPK];
    __shared__ float rsum[TOPK];
    __shared__ int   bestIdx;

    const int q = blockIdx.x;
    const int tid = threadIdx.x;

    // async copy: 2048 floats + 2048 ints, 32 B per lane per array (2 x b128)
    {
        const float* gs = part_score + (size_t)q * NCAND + tid * 8;
        const int*   gi = part_idx   + (size_t)q * NCAND + tid * 8;
        unsigned ls = LDS_OFF(&cs[tid * 8]);
        unsigned li = LDS_OFF(&ci[tid * 8]);
        asm volatile("global_load_async_to_lds_b128 %0, %1, off"
                     :: "v"(ls), "v"((unsigned long long)(uintptr_t)gs) : "memory");
        asm volatile("global_load_async_to_lds_b128 %0, %1, off"
                     :: "v"(ls + 16u), "v"((unsigned long long)(uintptr_t)(gs + 4)) : "memory");
        asm volatile("global_load_async_to_lds_b128 %0, %1, off"
                     :: "v"(li), "v"((unsigned long long)(uintptr_t)gi) : "memory");
        asm volatile("global_load_async_to_lds_b128 %0, %1, off"
                     :: "v"(li + 16u), "v"((unsigned long long)(uintptr_t)(gi + 4)) : "memory");
        asm volatile("s_wait_asynccnt 0x0" ::: "memory");
    }

    for (int k = 0; k < TOPK; ++k) {
        __syncthreads();
        float bm = NEG_INF; int bp = 0;
        for (int i = tid; i < NCAND; i += 256)
            if (cs[i] > bm) { bm = cs[i]; bp = i; }
        rmax[tid] = bm; rpos[tid] = bp;
        __syncthreads();
        for (int s = 128; s > 0; s >>= 1) {
            if (tid < s && rmax[tid + s] > rmax[tid]) {
                rmax[tid] = rmax[tid + s]; rpos[tid] = rpos[tid + s];
            }
            __syncthreads();
        }
        if (tid == 0) { const int p = rpos[0]; win[k] = ci[p]; cs[p] = NEG_INF; }
    }
    __syncthreads();
    if (tid < TOPK) {
        const float* mrow = memories + (size_t)win[tid] * MEMD;
        float s = 0.f;
        #pragma unroll
        for (int j = 0; j < RETD; ++j) s += mrow[OBSD + ACTD + j];
        rsum[tid] = s;
    }
    __syncthreads();
    if (tid == 0) {
        float bb = rsum[0]; int bj = 0;
        for (int j = 1; j < TOPK; ++j) if (rsum[j] > bb) { bb = rsum[j]; bj = j; }
        bestIdx = win[bj];
    }
    __syncthreads();
    if (tid < ACTD)
        best_act[q * ACTD + tid] = memories[(size_t)bestIdx * MEMD + OBSD + tid];
}

// ---------------------------------------------------------------------------
// Phase 3: logits = tanh(concat(tanh(obs@W_obs+b_obs), best_act) @ W_out + b_out)
// One block of 256 threads (8 waves x 16 queries), bf16 WMMA, K=80 padded to 96.
// ---------------------------------------------------------------------------
__global__ __launch_bounds__(256)
void p3_mlp(const float* __restrict__ obs,
            const float* __restrict__ best_act,
            const float* __restrict__ W_obs, const float* __restrict__ b_obs,
            const float* __restrict__ W_out, const float* __restrict__ b_out,
            float* __restrict__ out)
{
    __shared__ __align__(32) __bf16 emb[8][16][96];  // [wave][query][embed|act|zero-pad]
    const int tid  = threadIdx.x;
    const int wave = tid >> 5;
    const int lane = tid & 31;
    const int lo   = lane & 15;
    const bool hiH = (lane >= 16);
    const int q0   = wave * 16;
    const int koff = hiH ? 16 : 0;
    const int mrow = hiH ? 8 : 0;

    // A fragments from obs
    v16bf a0, a1;
    {
        const float* orow = obs + (size_t)(q0 + lo) * OBSD;
        const int kb = hiH ? 16 : 0;
        #pragma unroll
        for (int e = 0; e < 8; ++e) {
            a0[e]     = (__bf16)orow[kb + e];
            a0[e + 8] = (__bf16)orow[8 + kb + e];
            a1[e]     = (__bf16)orow[32 + kb + e];
            a1[e + 8] = (__bf16)orow[40 + kb + e];
        }
    }

    // GEMM1: embed = tanh(obs @ W_obs + b_obs), 4 column tiles of 16
    for (int nt = 0; nt < 4; ++nt) {
        const int n = nt * 16 + lo;
        v16bf b0, b1;
        #pragma unroll
        for (int e = 0; e < 16; ++e) {
            b0[e] = (__bf16)W_obs[(koff + e) * EMB + n];
            b1[e] = (__bf16)W_obs[(32 + koff + e) * EMB + n];
        }
        v8f c = {};
        c = __builtin_amdgcn_wmma_f32_16x16x32_bf16(false, a0, false, b0, (short)0, c, false, false);
        c = __builtin_amdgcn_wmma_f32_16x16x32_bf16(false, a1, false, b1, (short)0, c, false, false);
        const float bias = b_obs[n];
        #pragma unroll
        for (int r = 0; r < 8; ++r)
            emb[wave][mrow + r][n] = (__bf16)tanhf(c[r] + bias);
    }
    // append best action and zero pad K 80..95
    if (!hiH) {
        #pragma unroll
        for (int j = 0; j < ACTD; ++j)
            emb[wave][lo][EMB + j] = (__bf16)best_act[(q0 + lo) * ACTD + j];
        #pragma unroll
        for (int j = 80; j < 96; ++j) emb[wave][lo][j] = (__bf16)0.f;
    }
    __syncthreads();

    // A fragments from emb rows (K = 96, last 16 are zeros)
    v16bf f0, f1, f2;
    {
        const __bf16* erow = &emb[wave][lo][0];
        const int kb = hiH ? 16 : 0;
        #pragma unroll
        for (int e = 0; e < 8; ++e) {
            f0[e] = erow[kb + e];      f0[e + 8] = erow[8 + kb + e];
            f1[e] = erow[32 + kb + e]; f1[e + 8] = erow[40 + kb + e];
            f2[e] = erow[64 + kb + e]; f2[e + 8] = erow[72 + kb + e];
        }
    }
    // GEMM2: logits = tanh(emb @ W_out + b_out)
    for (int nt = 0; nt < 4; ++nt) {
        const int n = nt * 16 + lo;
        v16bf w0, w1, w2;
        #pragma unroll
        for (int e = 0; e < 16; ++e) {
            w0[e] = (__bf16)W_out[(koff + e) * OUTD + n];
            w1[e] = (__bf16)W_out[(32 + koff + e) * OUTD + n];
            const int k2 = 64 + koff + e;
            w2[e] = (k2 < 80) ? (__bf16)W_out[k2 * OUTD + n] : (__bf16)0.f;
        }
        v8f c = {};
        c = __builtin_amdgcn_wmma_f32_16x16x32_bf16(false, f0, false, w0, (short)0, c, false, false);
        c = __builtin_amdgcn_wmma_f32_16x16x32_bf16(false, f1, false, w1, (short)0, c, false, false);
        c = __builtin_amdgcn_wmma_f32_16x16x32_bf16(false, f2, false, w2, (short)0, c, false, false);
        const float bias = b_out[n];
        #pragma unroll
        for (int r = 0; r < 8; ++r)
            out[(size_t)(q0 + mrow + r) * OUTD + n] = tanhf(c[r] + bias);
    }
}

// ---------------------------------------------------------------------------
extern "C" void kernel_launch(void* const* d_in, const int* in_sizes, int n_in,
                              void* d_out, int out_size, void* d_ws, size_t ws_size,
                              hipStream_t stream) {
    (void)in_sizes; (void)n_in; (void)out_size; (void)ws_size;
    const float* obs      = (const float*)d_in[0];
    const float* memories = (const float*)d_in[1];
    const float* W_obs    = (const float*)d_in[2];
    const float* b_obs    = (const float*)d_in[3];
    const float* W_out    = (const float*)d_in[4];
    const float* b_out    = (const float*)d_in[5];
    float* out = (float*)d_out;

    // workspace layout: partial top-16 lists (scores, indices) + best actions
    float* part_score = (float*)d_ws;                                   // NB*NCH*TOPK f32 (1 MB)
    int*   part_idx   = (int*)(part_score + (size_t)NB * NCH * TOPK);   // 1 MB
    float* best_act   = (float*)(part_idx + (size_t)NB * NCH * TOPK);   // NB*ACTD f32 (8 KB)

    p1_scores_topk<<<NCH, 256, 0, stream>>>(obs, memories, part_score, part_idx);
    p2_select<<<NB, 256, 0, stream>>>(memories, part_score, part_idx, best_act);
    p3_mlp<<<1, 256, 0, stream>>>(obs, best_act, W_obs, b_obs, W_out, b_out, out);
}
